// SAN_46961172414543
// MI455X (gfx1250) — matrix operand order repaired and checked by reading, not compiled, not run on hardware
//
#include <hip/hip_runtime.h>
#include <hip/hip_bf16.h>

// ---------------------------------------------------------------------------
// SAN graph-transformer forward for MI455X (gfx1250, wave32, WMMA).
//
// Roofline: ~40 GFLOP of GEMM is trivial for the WMMA engines; the run is
// bound by the 1.6M-edge HBM streams (e_emb / Ee kept in f16, ~1.7 GB ->
// ~70us @ 23.3 TB/s). Node tensors (25.6 MB) stay in the 192 MB L2, so the
// per-edge gathers and f32 atomic scatter are on-chip.
//
// GEMM scheme: W staged once per block into LDS in B-fragment layout, then
// hoisted into registers (<=16 x v16h) per wave; each wave computes MT 16xN
// row strips so the steady-state loop is global_load_b128 -> cvt ->
// v_wmma_f32_16x16x32_f16 x NT with no LDS waits. K is compile-time, so
// everything unrolls and fragment arrays stay in registers.
// ---------------------------------------------------------------------------

typedef _Float16 half_t;
typedef __attribute__((ext_vector_type(16))) _Float16 v16h;
typedef __attribute__((ext_vector_type(8)))  _Float16 v8h;
typedef __attribute__((ext_vector_type(8)))  float    v8f;

#define DEVINL static __device__ __forceinline__

// ---- A fragment (16x32 f16), CDNA5 ISA 7.12.2 layout -----------------------
// lanes 0-15: rows M=0..15, K slices {kb..kb+7, kb+16..kb+23}, kb=0;
// lanes 16-31: same rows, kb=8. Branch-free; K==16 zero-pads the high half.
template <typename AT, int K>
DEVINL v16h load_a_frag(const AT* __restrict__ A, int row0, int k0, int lane) {
  const int m  = row0 + (lane & 15);
  const int kb = k0 + ((lane >> 4) << 3);
  const AT* __restrict__ row = A + (size_t)m * K;
  v16h a;
  if constexpr (sizeof(AT) == 4) {
    const float4 x0 = *(const float4*)(row + kb);
    const float4 x1 = *(const float4*)(row + kb + 4);
    a[0] = (half_t)x0.x; a[1] = (half_t)x0.y; a[2] = (half_t)x0.z; a[3] = (half_t)x0.w;
    a[4] = (half_t)x1.x; a[5] = (half_t)x1.y; a[6] = (half_t)x1.z; a[7] = (half_t)x1.w;
    if constexpr (K == 16) {
#pragma unroll
      for (int j = 8; j < 16; ++j) a[j] = (half_t)0.f;   // k >= 16 for every lane
    } else {
      const float4 y0 = *(const float4*)(row + kb + 16);
      const float4 y1 = *(const float4*)(row + kb + 20);
      a[8]  = (half_t)y0.x; a[9]  = (half_t)y0.y; a[10] = (half_t)y0.z; a[11] = (half_t)y0.w;
      a[12] = (half_t)y1.x; a[13] = (half_t)y1.y; a[14] = (half_t)y1.z; a[15] = (half_t)y1.w;
    }
  } else {
    const v8h lo = *(const v8h*)(row + kb);
#pragma unroll
    for (int j = 0; j < 8; ++j) a[j] = lo[j];
    if constexpr (K == 16) {
#pragma unroll
      for (int j = 8; j < 16; ++j) a[j] = (half_t)0.f;
    } else {
      const v8h hi = *(const v8h*)(row + kb + 16);
#pragma unroll
      for (int j = 0; j < 8; ++j) a[j + 8] = hi[j];
    }
  }
  return a;
}

// ---- GEMM: C[M,N] = act(A[M,K] @ W[K,N] + bias) ----------------------------
// Block = 256 threads = 8 waves; each wave computes MT consecutive 16xN
// strips. W fragments live in LDS only transiently; each wave hoists them
// into registers before the M loop.
template <typename AT, typename CT, int N, int K, int MT>
__global__ __launch_bounds__(256) void gemm_wmma_kernel(
    const AT* __restrict__ A, const half_t* __restrict__ W,
    const float* __restrict__ bias, CT* __restrict__ C,
    int M, int relu)
{
  constexpr int NT = N / 16;
  constexpr int KS = (K + 31) / 32;       // k-steps
  constexpr int NF = KS * NT;             // B fragments per wave (<= 16)
  __shared__ __align__(32) half_t smem[NF * 512];

  const int lane = threadIdx.x & 31;
  const int wave = threadIdx.x >> 5;

  // Cooperative fill of W fragments (zero-pad k >= K for K==16).
  for (int i = threadIdx.x; i < NF * 512; i += 256) {
    const int j  = i & 15;
    const int l  = (i >> 4) & 31;
    const int ft = i >> 9;                 // ki*NT + t
    const int t  = ft % NT;
    const int ki = ft / NT;
    const int k  = (ki << 5) + ((l >> 4) << 3) + (j & 7) + ((j >> 3) << 4);
    const int n  = t * 16 + (l & 15);
    smem[i] = (k < K) ? W[(size_t)k * N + n] : (half_t)0.f;
  }
  __syncthreads();

  // Hoist all B fragments into registers (constant indices after unroll).
  v16h breg[NF];
#pragma unroll
  for (int f = 0; f < NF; ++f)
    breg[f] = *(const v16h*)(smem + ((f << 5) + lane) * 16);

  const long tile0 = ((long)blockIdx.x * 8 + wave) * MT;
#pragma unroll
  for (int mt = 0; mt < MT; ++mt) {
    const long rt = tile0 + mt;
    if (rt * 16 >= M) break;               // wave-uniform -> EXEC all-ones
    const int row0 = (int)rt * 16;

    v8f c[NT];
#pragma unroll
    for (int t = 0; t < NT; ++t) c[t] = (v8f){};

#pragma unroll
    for (int ki = 0; ki < KS; ++ki) {
      const v16h a = load_a_frag<AT, K>(A, row0, ki << 5, lane);
#pragma unroll
      for (int t = 0; t < NT; ++t)
        c[t] = __builtin_amdgcn_wmma_f32_16x16x32_f16(
            false, a, false, breg[ki * NT + t], (short)0, c[t], false, false);
    }

    // C/D layout: VGPR v -> M = row0 + v + (lane>=16 ? 8 : 0), N = t*16+(lane&15)
    const int mbase = row0 + ((lane >> 4) << 3);
    const int ncol  = lane & 15;
#pragma unroll
    for (int t = 0; t < NT; ++t) {
      const int n = t * 16 + ncol;
      const float bn = bias ? bias[n] : 0.f;
#pragma unroll
      for (int v = 0; v < 8; ++v) {
        float val = c[t][v] + bn;
        if (relu) val = fmaxf(val, 0.f);
        C[(size_t)(mbase + v) * N + n] = (CT)val;
      }
    }
  }
}

// ---- f32 -> f16 weight conversion -----------------------------------------
__global__ void f32_to_f16_kernel(const float* __restrict__ s, half_t* __restrict__ d, int n) {
  int i = blockIdx.x * 256 + threadIdx.x;
  if (i < n) d[i] = (half_t)s[i];
}

// ---- Edge attention: 8 threads per edge (one per head, 8 dims each). -------
// score = clip(sum_d K[src]*Q[dst]*Ee / sqrt(8), +-5); s = exp(score)
// wV[dst] += V[src]*s ; z[dst] += s   (f32 atomics, L2-resident targets)
__global__ __launch_bounds__(256) void edge_attn_kernel(
    const float* __restrict__ Kf, const float* __restrict__ Qf,
    const half_t* __restrict__ Ee, const float* __restrict__ Vf,
    const int* __restrict__ src, const int* __restrict__ dst,
    float* __restrict__ wV, float* __restrict__ z, int nE)
{
  const long t = (long)blockIdx.x * blockDim.x + threadIdx.x;
  const long e = t >> 3;
  const int  h = (int)(t & 7);
  if (e >= nE) return;
  const int s = src[e];
  const int d = dst[e];

  const float4* kp = (const float4*)(Kf + (size_t)s * 64 + h * 8);
  const float4* qp = (const float4*)(Qf + (size_t)d * 64 + h * 8);
  const float4* vp = (const float4*)(Vf + (size_t)s * 64 + h * 8);
  const v8h     ev = *(const v8h*)(Ee + (size_t)e * 64 + h * 8);

  const float4 k0 = kp[0], k1 = kp[1];
  const float4 q0 = qp[0], q1 = qp[1];
  const float scale = 0.35355339059327373f;  // 1/sqrt(8)

  float acc = k0.x * q0.x * (float)ev[0] + k0.y * q0.y * (float)ev[1]
            + k0.z * q0.z * (float)ev[2] + k0.w * q0.w * (float)ev[3]
            + k1.x * q1.x * (float)ev[4] + k1.y * q1.y * (float)ev[5]
            + k1.z * q1.z * (float)ev[6] + k1.w * q1.w * (float)ev[7];
  acc *= scale;
  acc = fminf(5.f, fmaxf(-5.f, acc));
  const float sc = __expf(acc);

  const float4 v0 = vp[0], v1 = vp[1];
  float* w = wV + (size_t)d * 64 + h * 8;
  atomicAdd(w + 0, v0.x * sc); atomicAdd(w + 1, v0.y * sc);
  atomicAdd(w + 2, v0.z * sc); atomicAdd(w + 3, v0.w * sc);
  atomicAdd(w + 4, v1.x * sc); atomicAdd(w + 5, v1.y * sc);
  atomicAdd(w + 6, v1.z * sc); atomicAdd(w + 7, v1.w * sc);
  atomicAdd(z + (size_t)d * 8 + h, sc);
}

// ---- a = wV / (z + 1e-6) in place -----------------------------------------
__global__ void attn_norm_kernel(float* __restrict__ wV, const float* __restrict__ z, int nNodes) {
  const long i = (long)blockIdx.x * blockDim.x + threadIdx.x;
  if (i >= (long)nNodes * 64) return;
  const long node = i >> 6;
  const int  head = (int)((i & 63) >> 3);
  wV[i] = wV[i] / (z[node * 8 + head] + 1e-6f);
}

// ---- out = LayerNorm(x [+ res]) * g + b ; one wave32 per 64-dim node row ----
__global__ __launch_bounds__(256) void add_ln_kernel(
    const float* __restrict__ x, const float* __restrict__ res,
    const float* __restrict__ g, const float* __restrict__ b,
    float* __restrict__ out, int nNodes)
{
  const int lane = threadIdx.x & 31;
  const int wave = threadIdx.x >> 5;
  const long node = (long)blockIdx.x * 8 + wave;
  if (node >= nNodes) return;
  const size_t base = (size_t)node * 64;
  float v0 = x[base + lane];
  float v1 = x[base + lane + 32];
  if (res) { v0 += res[base + lane]; v1 += res[base + lane + 32]; }
  float s = v0 + v1;
#pragma unroll
  for (int m = 16; m > 0; m >>= 1) s += __shfl_xor(s, m, 32);
  const float mu = s * (1.f / 64.f);
  const float d0 = v0 - mu, d1 = v1 - mu;
  float vs = d0 * d0 + d1 * d1;
#pragma unroll
  for (int m = 16; m > 0; m >>= 1) vs += __shfl_xor(vs, m, 32);
  const float inv = rsqrtf(vs * (1.f / 64.f) + 1e-5f);
  out[base + lane]      = d0 * inv * g[lane]      + b[lane];
  out[base + lane + 32] = d1 * inv * g[lane + 32] + b[lane + 32];
}

// ---- mean readout stage 1: column sums into acc[64] ------------------------
__global__ void mean_reduce_kernel(const float* __restrict__ h, float* __restrict__ acc, int nNodes) {
  __shared__ float s[64];
  if (threadIdx.x < 64) s[threadIdx.x] = 0.f;
  __syncthreads();
  const long total  = (long)nNodes * 64;
  const long stride = (long)gridDim.x * blockDim.x;      // multiple of 64 -> column fixed
  float local = 0.f;
  for (long i = (long)blockIdx.x * blockDim.x + threadIdx.x; i < total; i += stride)
    local += h[i];
  atomicAdd(&s[threadIdx.x & 63], local);                // LDS float atomic
  __syncthreads();
  if (threadIdx.x < 64) atomicAdd(&acc[threadIdx.x], s[threadIdx.x]);
}

// ---- readout MLP 64->32->16->3 (single tiny block) -------------------------
__global__ void readout_mlp_kernel(
    const float* __restrict__ acc,
    const float* __restrict__ w0, const float* __restrict__ b0,
    const float* __restrict__ w1, const float* __restrict__ b1,
    const float* __restrict__ w2, const float* __restrict__ b2,
    float* __restrict__ out, float invN)
{
  __shared__ float x[64], y[32], u[16];
  const int t = threadIdx.x;
  x[t] = acc[t] * invN;
  __syncthreads();
  if (t < 32) { float s = b0[t]; for (int i = 0; i < 64; ++i) s += x[i] * w0[i * 32 + t]; y[t] = fmaxf(s, 0.f); }
  __syncthreads();
  if (t < 16) { float s = b1[t]; for (int i = 0; i < 32; ++i) s += y[i] * w1[i * 16 + t]; u[t] = fmaxf(s, 0.f); }
  __syncthreads();
  if (t < 3)  { float s = b2[t]; for (int i = 0; i < 16; ++i) s += u[i] * w2[i * 3 + t];  out[t] = s; }
}

// ---------------------------------------------------------------------------
// Host-side orchestration
// ---------------------------------------------------------------------------
static const int N_NODES = 100000;
static const int N_EDGES = 1600000;

template <typename AT, typename CT, int N, int K, int MT>
static void gemm(const AT* A, const half_t* W, const float* bias, CT* C,
                 int M, int relu, hipStream_t s) {
  const long tiles = M / 16;
  const int blocks = (int)((tiles + 8 * MT - 1) / (8 * MT));
  gemm_wmma_kernel<AT, CT, N, K, MT><<<blocks, 256, 0, s>>>(A, W, bias, C, M, relu);
}

extern "C" void kernel_launch(void* const* d_in, const int* in_sizes, int n_in,
                              void* d_out, int out_size, void* d_ws, size_t ws_size,
                              hipStream_t stream)
{
  (void)in_sizes; (void)n_in; (void)out_size; (void)ws_size;

  // Inputs: setup_inputs() order, then params in jax pytree flatten order
  // (dict keys alphabetical; None biases dropped).
  const float* h_in  = (const float*)d_in[0];
  const float* e_in  = (const float*)d_in[1];
  const int*   src   = (const int*)d_in[2];
  const int*   dst   = (const int*)d_in[3];
  const float* embEb = (const float*)d_in[4];
  const float* embEw = (const float*)d_in[5];   // [16,64]
  const float* embHb = (const float*)d_in[6];
  const float* embHw = (const float*)d_in[7];   // [64,64]

  struct Layer { const float *Ew,*Kw,*Ob,*Ow,*Qw,*Vw,*f1b,*f1w,*f2b,*f2w,*ln1b,*ln1g,*ln2b,*ln2g; };
  Layer L[2];
  for (int l = 0; l < 2; ++l) {
    const int p = 8 + l * 14;   // keys: E,K,O,Q,V,ffn1,ffn2,ln1,ln2 (b before w)
    L[l].Ew   = (const float*)d_in[p + 0];
    L[l].Kw   = (const float*)d_in[p + 1];
    L[l].Ob   = (const float*)d_in[p + 2];
    L[l].Ow   = (const float*)d_in[p + 3];
    L[l].Qw   = (const float*)d_in[p + 4];
    L[l].Vw   = (const float*)d_in[p + 5];
    L[l].f1b  = (const float*)d_in[p + 6];
    L[l].f1w  = (const float*)d_in[p + 7];   // [64,128]
    L[l].f2b  = (const float*)d_in[p + 8];
    L[l].f2w  = (const float*)d_in[p + 9];   // [128,64]
    L[l].ln1b = (const float*)d_in[p + 10];
    L[l].ln1g = (const float*)d_in[p + 11];
    L[l].ln2b = (const float*)d_in[p + 12];
    L[l].ln2g = (const float*)d_in[p + 13];
  }
  const float* mB0 = (const float*)d_in[36]; const float* mW0 = (const float*)d_in[37];
  const float* mB1 = (const float*)d_in[38]; const float* mW1 = (const float*)d_in[39];
  const float* mB2 = (const float*)d_in[40]; const float* mW2 = (const float*)d_in[41];

  // ---- workspace: [0, 256KB) f16 weights; then f32 buffers; f16 edge bufs --
  half_t* wh = (half_t*)d_ws;
  size_t whOff = 0;
  auto cvt = [&](const float* w, int n) -> half_t* {
    half_t* p = wh + whOff;
    f32_to_f16_kernel<<<(n + 255) / 256, 256, 0, stream>>>(w, p, n);
    whOff += (size_t)n;
    return p;
  };
  half_t* WembH = cvt(embHw, 64 * 64);
  half_t* WembE = cvt(embEw, 16 * 64);
  half_t *WQ[2], *WK[2], *WV[2], *WE[2], *WO[2], *WF1[2], *WF2[2];
  for (int l = 0; l < 2; ++l) {
    WQ[l]  = cvt(L[l].Qw, 64 * 64);
    WK[l]  = cvt(L[l].Kw, 64 * 64);
    WV[l]  = cvt(L[l].Vw, 64 * 64);
    WE[l]  = cvt(L[l].Ew, 64 * 64);
    WO[l]  = cvt(L[l].Ow, 64 * 64);
    WF1[l] = cvt(L[l].f1w, 64 * 128);
    WF2[l] = cvt(L[l].f2w, 128 * 64);
  }

  float* fp = (float*)((char*)d_ws + (256 << 10));
  auto falloc = [&](size_t n) -> float* { float* p = fp; fp += n; return p; };
  float* hin = falloc((size_t)N_NODES * 64);   // layer input / final node features
  float* qb  = falloc((size_t)N_NODES * 64);
  float* kb  = falloc((size_t)N_NODES * 64);
  float* vb  = falloc((size_t)N_NODES * 64);
  float* wv  = falloc((size_t)N_NODES * 64);   // segment-sum accumulator -> a
  float* ob  = falloc((size_t)N_NODES * 64);   // O / ffn2 output
  float* hln = falloc((size_t)N_NODES * 64);   // post-LN1
  float* tb  = falloc((size_t)N_NODES * 128);  // ffn hidden
  float* zb  = falloc((size_t)N_NODES * 8);    // softmax denominators
  float* acc = falloc(64);                     // readout column sums
  half_t* e0  = (half_t*)fp;                   // edge embeddings (f16, HBM-stream)
  half_t* Eeb = e0 + (size_t)N_EDGES * 64;     // per-layer edge projection (f16)

  // ---- embeddings ----------------------------------------------------------
  gemm<float, float,  64, 64, 4>(h_in, WembH, embHb, hin, N_NODES, 0, stream);
  gemm<float, half_t, 64, 16, 4>(e_in, WembE, embEb, e0,  N_EDGES, 0, stream);

  // ---- transformer layers --------------------------------------------------
  for (int l = 0; l < 2; ++l) {
    gemm<float,  float,  64, 64, 4>(hin, WQ[l], nullptr, qb,  N_NODES, 0, stream);
    gemm<float,  float,  64, 64, 4>(hin, WK[l], nullptr, kb,  N_NODES, 0, stream);
    gemm<float,  float,  64, 64, 4>(hin, WV[l], nullptr, vb,  N_NODES, 0, stream);
    gemm<half_t, half_t, 64, 64, 4>(e0,  WE[l], nullptr, Eeb, N_EDGES, 0, stream);

    hipMemsetAsync(wv, 0, (size_t)N_NODES * 64 * sizeof(float), stream);
    hipMemsetAsync(zb, 0, (size_t)N_NODES * 8 * sizeof(float), stream);

    edge_attn_kernel<<<(N_EDGES * 8) / 256, 256, 0, stream>>>(
        kb, qb, Eeb, vb, src, dst, wv, zb, N_EDGES);
    attn_norm_kernel<<<(N_NODES * 64) / 256, 256, 0, stream>>>(wv, zb, N_NODES);

    gemm<float, float, 64, 64, 4>(wv, WO[l], L[l].Ob, ob, N_NODES, 0, stream);
    // residual only on layer 0 (i < N_LAYERS-1)
    add_ln_kernel<<<N_NODES / 8, 256, 0, stream>>>(
        ob, (l == 0) ? hin : nullptr, L[l].ln1g, L[l].ln1b, hln, N_NODES);

    gemm<float, float, 128, 64, 2>(hln, WF1[l], L[l].f1b, tb, N_NODES, 1, stream);
    gemm<float, float, 64, 128, 2>(tb,  WF2[l], L[l].f2b, ob, N_NODES, 0, stream);
    add_ln_kernel<<<N_NODES / 8, 256, 0, stream>>>(
        ob, hln, L[l].ln2g, L[l].ln2b, hin, N_NODES);
  }

  // ---- readout: mean over nodes, then 64->32->16->3 MLP --------------------
  hipMemsetAsync(acc, 0, 64 * sizeof(float), stream);
  mean_reduce_kernel<<<1024, 256, 0, stream>>>(hin, acc, N_NODES);
  readout_mlp_kernel<<<1, 64, 0, stream>>>(acc, mW0, mB0, mW1, mB1, mW2, mB2,
                                           (float*)d_out, 1.0f / (float)N_NODES);
}
